// SUNet_37477884625187
// MI455X (gfx1250) — matrix operand order repaired and checked by reading, not compiled, not run on hardware
//
#include <hip/hip_runtime.h>
#include <math.h>

// ---------------------------------------------------------------------------
// SUNet (Swin U-Net) forward for MI455X (gfx1250, wave32).
// All dense GEMMs (qkv/proj/fc1/fc2/1x1 convs/patch-merge/attention) run on
// v_wmma_f32_16x16x32_f16: f16 inputs, f32 accumulation. The generic GEMM is
// register-blocked 2x2 (32x32 output per wave -> 4 WMMAs per fragment-load
// group). Window partition, cyclic shift and reverse-shift are folded into
// index math in the attention kernels. Head dim is 32; window tokens = 64.
// ---------------------------------------------------------------------------

typedef __attribute__((ext_vector_type(16))) _Float16 v16h;
typedef __attribute__((ext_vector_type(8)))  float    v8f;

#define ACT_NONE  0
#define ACT_GELU  1
#define ACT_PRELU 2

__device__ __forceinline__ float gelu_exact(float x) {
    return 0.5f * x * (1.0f + erff(x * 0.70710678118654752f));
}

__device__ __forceinline__ v16h load_frag_a(const float* p0, const float* p1) {
    v16h a;
#pragma unroll
    for (int e = 0; e < 8; ++e) {
        a[e]     = (_Float16)p0[e];
        a[8 + e] = (_Float16)p1[e];
    }
    return a;
}
__device__ __forceinline__ v16h load_frag_b(const float* p0, const float* p1) {
    v16h b;
#pragma unroll
    for (int e = 0; e < 8; ++e) {
        b[e]     = (_Float16)p0[e];
        b[8 + e] = (_Float16)p1[e];
    }
    return b;
}

// ------------------------------ generic GEMM -------------------------------
// C[M,N] = act(A[M,K] @ W[N,K]^T + bias) (+ residual)
// Each wave computes a 32x32 output block (2x2 WMMA tiles); 4 waves per block
// cover 128 rows. M must be a multiple of 16; N a multiple of 32 (true for
// every GEMM in this network).
template <int ACT>
__launch_bounds__(128)
__global__ void gemm_wmma(const float* __restrict__ A, const float* __restrict__ Wt,
                          const float* __restrict__ bias, const float* __restrict__ res,
                          const float* __restrict__ actp, float* __restrict__ C,
                          int M, int N, int K)
{
    const int lane = threadIdx.x;            // 0..31
    const int wv   = threadIdx.y;            // 0..3
    const int tile_n = blockIdx.x;           // 32-col block
    const int tile_m = blockIdx.y * 4 + wv;  // 32-row block
    if (tile_m * 32 >= M) return;
    const int lm = lane & 15;
    const int hi = lane >> 4;

    const float* Arow0 = A  + (size_t)(tile_m * 32 + lm) * K;
    const float* Arow1 = Arow0 + (size_t)16 * K;
    const float* Wrow0 = Wt + (size_t)(tile_n * 32 + lm) * K;
    const float* Wrow1 = Wrow0 + (size_t)16 * K;

    v8f acc00 = {}, acc01 = {}, acc10 = {}, acc11 = {};
    for (int kb = 0; kb < K; kb += 32) {
        const int ka0 = kb + hi * 8;
        const int ka1 = kb + 16 + hi * 8;
        const v16h a0 = load_frag_a(Arow0 + ka0, Arow0 + ka1);
        const v16h a1 = load_frag_a(Arow1 + ka0, Arow1 + ka1);
        const v16h b0 = load_frag_b(Wrow0 + ka0, Wrow0 + ka1);
        const v16h b1 = load_frag_b(Wrow1 + ka0, Wrow1 + ka1);
        acc00 = __builtin_amdgcn_wmma_f32_16x16x32_f16(false, a0, false, b0, (short)0, acc00, false, false);
        acc01 = __builtin_amdgcn_wmma_f32_16x16x32_f16(false, a0, false, b1, (short)0, acc01, false, false);
        acc10 = __builtin_amdgcn_wmma_f32_16x16x32_f16(false, a1, false, b0, (short)0, acc10, false, false);
        acc11 = __builtin_amdgcn_wmma_f32_16x16x32_f16(false, a1, false, b1, (short)0, acc11, false, false);
    }

    const int col0 = tile_n * 32 + lm;
    const int col1 = col0 + 16;
    const float bv0 = bias ? bias[col0] : 0.0f;
    const float bv1 = bias ? bias[col1] : 0.0f;
    const float ap  = (ACT == ACT_PRELU) ? actp[0] : 0.0f;

    const int row_lim = M;   // rows within last block may exceed M only if M%32
#pragma unroll
    for (int ti = 0; ti < 2; ++ti) {
        const v8f* accs0 = (ti == 0) ? &acc00 : &acc10;
        const v8f* accs1 = (ti == 0) ? &acc01 : &acc11;
#pragma unroll
        for (int v = 0; v < 8; ++v) {
            const int row = tile_m * 32 + ti * 16 + hi * 8 + v;
            if (row >= row_lim) continue;
            float x0 = (*accs0)[v] + bv0;
            float x1 = (*accs1)[v] + bv1;
            if (ACT == ACT_GELU)  { x0 = gelu_exact(x0); x1 = gelu_exact(x1); }
            if (ACT == ACT_PRELU) { x0 = (x0 >= 0.0f) ? x0 : ap * x0;
                                    x1 = (x1 >= 0.0f) ? x1 : ap * x1; }
            if (res) {
                x0 += res[(size_t)row * N + col0];
                x1 += res[(size_t)row * N + col1];
            }
            C[(size_t)row * N + col0] = x0;
            C[(size_t)row * N + col1] = x1;
        }
    }
}

// ------------------------------ LayerNorm ----------------------------------
__launch_bounds__(256)
__global__ void layernorm_k(const float* __restrict__ X, const float* __restrict__ g,
                            const float* __restrict__ be, float* __restrict__ Y,
                            int rows, int C)
{
    const int row = blockIdx.x * 8 + (threadIdx.x >> 5);
    if (row >= rows) return;
    const int lane = threadIdx.x & 31;
    const float* x = X + (size_t)row * C;
    float s = 0.0f, ss = 0.0f;
    for (int c = lane; c < C; c += 32) { float v = x[c]; s += v; ss += v * v; }
#pragma unroll
    for (int t = 16; t; t >>= 1) { s += __shfl_xor(s, t, 32); ss += __shfl_xor(ss, t, 32); }
    const float mean = s / (float)C;
    const float var  = ss / (float)C - mean * mean;
    const float rstd = rsqrtf(var + 1e-5f);
    float* y = Y + (size_t)row * C;
    for (int c = lane; c < C; c += 32) y[c] = (x[c] - mean) * rstd * g[c] + be[c];
}

// ----------------------- windowed attention: scores ------------------------
__device__ __forceinline__ int sw_region(int c, int D, int shift) {
    return (c < D - 8) ? 0 : ((c < D - shift) ? 1 : 2);
}

// S[winHead, 64, 64] = q*k^T/sqrt(32) + relpos_bias + shift_mask
__launch_bounds__(256)
__global__ void attn_scores(const float* __restrict__ qkv, const float* __restrict__ rpb,
                            float* __restrict__ S,
                            int H, int W, int C, int heads, int shift, int nW)
{
    const int wh   = blockIdx.x;
    const int head = wh % heads;
    const int w    = wh / heads;
    const int b    = w / nW;
    const int win  = w % nW;
    const int wpr  = W >> 3;
    const int wy   = (win / wpr) << 3;
    const int wx   = (win % wpr) << 3;
    const int lane = threadIdx.x & 31;
    const int wv   = threadIdx.x >> 5;
    const int hi   = lane >> 4, lm = lane & 15;
    const float scale = 0.17677669529663687f;   // 32^-0.5
    float* Sout = S + (size_t)wh * 64 * 64;

    for (int iter = 0; iter < 2; ++iter) {
        const int t  = wv + iter * 8;       // 0..15 tiles (4x4)
        const int tm = t >> 2, tn = t & 3;
        // A = Q rows
        const int qtok = tm * 16 + lm;
        int qy = wy + (qtok >> 3) + shift; if (qy >= H) qy -= H;
        int qx = wx + (qtok & 7)  + shift; if (qx >= W) qx -= W;
        const float* qp = qkv + ((size_t)(b * H + qy) * W + qx) * (3 * C) + head * 32;
        // B = K^T columns
        const int ktok = tn * 16 + lm;
        int ky = wy + (ktok >> 3) + shift; if (ky >= H) ky -= H;
        int kx = wx + (ktok & 7)  + shift; if (kx >= W) kx -= W;
        const float* kp = qkv + ((size_t)(b * H + ky) * W + kx) * (3 * C) + C + head * 32;

        v16h a, bb;
#pragma unroll
        for (int e = 0; e < 8; ++e) {
            a[e]     = (_Float16)(qp[hi * 8 + e] * scale);
            a[8 + e] = (_Float16)(qp[16 + hi * 8 + e] * scale);
        }
#pragma unroll
        for (int e = 0; e < 16; ++e) bb[e] = (_Float16)kp[(e < 8) ? (hi * 8 + e) : (16 + hi * 8 + (e - 8))];
        v8f acc = {};
        acc = __builtin_amdgcn_wmma_f32_16x16x32_f16(false, a, false, bb,
                                                     (short)0, acc, false, false);

        const int m  = tn * 16 + lm;
        const int my = m >> 3, mx = m & 7;
#pragma unroll
        for (int v = 0; v < 8; ++v) {
            const int n  = tm * 16 + hi * 8 + v;
            const int ny = n >> 3, nx = n & 7;
            const int idx = (ny - my + 7) * 15 + (nx - mx + 7);
            float val = acc[v] + rpb[idx * heads + head];
            if (shift > 0) {
                const int rn = sw_region(wy + ny, H, shift) * 3 + sw_region(wx + nx, W, shift);
                const int rm = sw_region(wy + my, H, shift) * 3 + sw_region(wx + mx, W, shift);
                if (rn != rm) val -= 100.0f;
            }
            Sout[n * 64 + m] = val;
        }
    }
}

// ----------------------------- softmax (row=64) ----------------------------
__launch_bounds__(256)
__global__ void softmax64(float* __restrict__ S, int rows)
{
    const int row = blockIdx.x * 8 + (threadIdx.x >> 5);
    if (row >= rows) return;
    const int lane = threadIdx.x & 31;
    float* p = S + (size_t)row * 64;
    float a = p[lane], b = p[lane + 32];
    float mx = fmaxf(a, b);
#pragma unroll
    for (int s = 16; s; s >>= 1) mx = fmaxf(mx, __shfl_xor(mx, s, 32));
    const float ea = __expf(a - mx), eb = __expf(b - mx);
    float sum = ea + eb;
#pragma unroll
    for (int s = 16; s; s >>= 1) sum += __shfl_xor(sum, s, 32);
    const float inv = 1.0f / sum;
    p[lane] = ea * inv;
    p[lane + 32] = eb * inv;
}

// -------------------------- attention: P @ V -------------------------------
// Writes per-head outputs directly back to (B, H*W, C) at un-shifted positions.
__launch_bounds__(256)
__global__ void attn_av(const float* __restrict__ qkv, const float* __restrict__ S,
                        float* __restrict__ O,
                        int H, int W, int C, int heads, int shift, int nW)
{
    const int wh   = blockIdx.x;
    const int head = wh % heads;
    const int w    = wh / heads;
    const int b    = w / nW;
    const int win  = w % nW;
    const int wpr  = W >> 3;
    const int wy   = (win / wpr) << 3;
    const int wx   = (win % wpr) << 3;
    const int lane = threadIdx.x & 31;
    const int wv   = threadIdx.x >> 5;    // 8 tiles: 4 (M) x 2 (d)
    const int tm   = wv >> 1, tn = wv & 1;
    const int hi   = lane >> 4, lm = lane & 15;

    const float* Srow = S + (size_t)wh * 64 * 64 + (size_t)(tm * 16 + lm) * 64;
    v8f acc = {};
    for (int kb = 0; kb < 64; kb += 32) {
        v16h a, bb;
#pragma unroll
        for (int e = 0; e < 8; ++e) {
            a[e]     = (_Float16)Srow[kb + hi * 8 + e];
            a[8 + e] = (_Float16)Srow[kb + 16 + hi * 8 + e];
        }
#pragma unroll
        for (int e = 0; e < 16; ++e) {
            const int m = kb + ((e < 8) ? (hi * 8 + e) : (16 + hi * 8 + (e - 8)));          // value token
            int my = wy + (m >> 3) + shift; if (my >= H) my -= H;
            int mx = wx + (m & 7)  + shift; if (mx >= W) mx -= W;
            bb[e] = (_Float16)qkv[((size_t)(b * H + my) * W + mx) * (3 * C)
                                  + 2 * C + head * 32 + tn * 16 + lm];
        }
        acc = __builtin_amdgcn_wmma_f32_16x16x32_f16(false, a, false, bb,
                                                     (short)0, acc, false, false);
    }
    const int col = head * 32 + tn * 16 + lm;
#pragma unroll
    for (int v = 0; v < 8; ++v) {
        const int n = tm * 16 + hi * 8 + v;
        int ny = wy + (n >> 3) + shift; if (ny >= H) ny -= H;
        int nx = wx + (n & 7)  + shift; if (nx >= W) nx -= W;
        O[((size_t)(b * H + ny) * W + nx) * C + col] = acc[v];
    }
}

// ---------------------- patch merge gather (2x2 concat) --------------------
__global__ void merge_gather(const float* __restrict__ X, float* __restrict__ Y,
                             int Bn, int H, int W, int C)
{
    const size_t i = (size_t)blockIdx.x * blockDim.x + threadIdx.x;
    const int H2 = H >> 1, W2 = W >> 1;
    const size_t total = (size_t)Bn * H2 * W2 * 4 * C;
    if (i >= total) return;
    const int c4 = (int)(i % (4 * C));
    size_t pix  = i / (4 * C);
    const int x2 = (int)(pix % W2); pix /= W2;
    const int y2 = (int)(pix % H2);
    const int b  = (int)(pix / H2);
    const int q  = c4 / C, c = c4 % C;
    const int dy = q & 1, dx = q >> 1;
    Y[i] = X[((size_t)(b * H + 2 * y2 + dy) * W + (2 * x2 + dx)) * C + c];
}

// ---------------------------- pixel shuffle --------------------------------
__global__ void pixel_shuffle_k(const float* __restrict__ X, float* __restrict__ Y,
                                int Bn, int H, int W, int Co, int r)
{
    const size_t i = (size_t)blockIdx.x * blockDim.x + threadIdx.x;
    const size_t total = (size_t)Bn * H * W * r * r * Co;
    if (i >= total) return;
    const int co = (int)(i % Co);
    size_t pix = i / Co;
    const int X2 = (int)(pix % (r * W)); pix /= (size_t)(r * W);
    const int Y2 = (int)(pix % (r * H));
    const int b  = (int)(pix / (r * H));
    const int h = Y2 / r, ii = Y2 % r, w0 = X2 / r, j = X2 % r;
    Y[i] = X[((size_t)(b * H + h) * W + w0) * (Co * r * r) + co * r * r + ii * r + j];
}

// ------------------------- bilinear upsample (x r) -------------------------
__global__ void bilinear_k(const float* __restrict__ X, float* __restrict__ Y,
                           int Bn, int H, int W, int C, int r)
{
    const size_t i = (size_t)blockIdx.x * blockDim.x + threadIdx.x;
    const size_t total = (size_t)Bn * H * W * r * r * C;
    if (i >= total) return;
    const int c = (int)(i % C);
    size_t pix = i / C;
    const int x = (int)(pix % (r * W)); pix /= (size_t)(r * W);
    const int y = (int)(pix % (r * H));
    const int b = (int)(pix / (r * H));
    const float fy = (y + 0.5f) / (float)r - 0.5f;
    const float fx = (x + 0.5f) / (float)r - 0.5f;
    int y0 = (int)floorf(fy), x0 = (int)floorf(fx);
    const float wy = fy - y0, wx = fx - x0;
    int y1 = y0 + 1, x1 = x0 + 1;
    y0 = min(max(y0, 0), H - 1); y1 = min(max(y1, 0), H - 1);
    x0 = min(max(x0, 0), W - 1); x1 = min(max(x1, 0), W - 1);
    const float* base = X + (size_t)b * H * W * C;
    const float v00 = base[((size_t)y0 * W + x0) * C + c];
    const float v01 = base[((size_t)y0 * W + x1) * C + c];
    const float v10 = base[((size_t)y1 * W + x0) * C + c];
    const float v11 = base[((size_t)y1 * W + x1) * C + c];
    Y[i] = v00 * (1 - wy) * (1 - wx) + v01 * (1 - wy) * wx
         + v10 * wy * (1 - wx) + v11 * wy * wx;
}

// ------------------------------ channel concat -----------------------------
__global__ void concat_k(const float* __restrict__ A, const float* __restrict__ B_,
                         float* __restrict__ Y, size_t rows, int Ca, int Cb)
{
    const size_t i = (size_t)blockIdx.x * blockDim.x + threadIdx.x;
    const size_t total = rows * (size_t)(Ca + Cb);
    if (i >= total) return;
    const int c = (int)(i % (Ca + Cb));
    const size_t row = i / (Ca + Cb);
    Y[i] = (c < Ca) ? A[row * Ca + c] : B_[row * Cb + (c - Ca)];
}

// --------------------- conv_first: (B,3,256,256)->(B,96,256,256) -----------
__launch_bounds__(256)
__global__ void conv3x3_first(const float* __restrict__ X, const float* __restrict__ Wt,
                              const float* __restrict__ bias, float* __restrict__ Y)
{
    const size_t i = (size_t)blockIdx.x * blockDim.x + threadIdx.x;
    if (i >= (size_t)4 * 96 * 256 * 256) return;
    const int x = (int)(i & 255);
    const int y = (int)((i >> 8) & 255);
    const int o = (int)((i >> 16) % 96);
    const int b = (int)(i / ((size_t)96 * 65536));
    float acc = bias[o];
    for (int c = 0; c < 3; ++c)
        for (int ky = 0; ky < 3; ++ky) {
            const int iy = y + ky - 1;
            if (iy < 0 || iy > 255) continue;
            for (int kx = 0; kx < 3; ++kx) {
                const int ix = x + kx - 1;
                if (ix < 0 || ix > 255) continue;
                acc += X[((size_t)(b * 3 + c) * 256 + iy) * 256 + ix]
                     * Wt[((o * 3 + c) * 3 + ky) * 3 + kx];
            }
        }
    Y[i] = acc;
}

// --------- patch embed: 4x4/4 conv NCHW -> channel-last tokens -------------
__launch_bounds__(256)
__global__ void patch_embed(const float* __restrict__ X, const float* __restrict__ Wt,
                            const float* __restrict__ bias, float* __restrict__ Y)
{
    const size_t i = (size_t)blockIdx.x * blockDim.x + threadIdx.x;
    if (i >= (size_t)4 * 4096 * 96) return;
    const int o = (int)(i % 96);
    const int l = (int)((i / 96) % 4096);
    const int b = (int)(i / ((size_t)96 * 4096));
    const int py = (l >> 6) << 2, px = (l & 63) << 2;
    float acc = bias[o];
    for (int c = 0; c < 96; ++c) {
        const float* xp = X + ((size_t)(b * 96 + c) * 256 + py) * 256 + px;
        const float* wp = Wt + ((size_t)(o * 96 + c)) * 16;
#pragma unroll
        for (int ky = 0; ky < 4; ++ky)
#pragma unroll
            for (int kx = 0; kx < 4; ++kx)
                acc += xp[ky * 256 + kx] * wp[ky * 4 + kx];
    }
    Y[i] = acc;
}

// ------- output conv: channel-last (B,256,256,96) -> NCHW (B,3,256,256) ----
__launch_bounds__(256)
__global__ void conv3x3_out(const float* __restrict__ X, const float* __restrict__ Wt,
                            float* __restrict__ Y)
{
    const size_t i = (size_t)blockIdx.x * blockDim.x + threadIdx.x;
    if (i >= (size_t)4 * 3 * 256 * 256) return;
    const int x = (int)(i & 255);
    const int y = (int)((i >> 8) & 255);
    const int o = (int)((i >> 16) % 3);
    const int b = (int)(i / ((size_t)3 * 65536));
    float acc = 0.0f;
    for (int ky = 0; ky < 3; ++ky) {
        const int iy = y + ky - 1;
        if (iy < 0 || iy > 255) continue;
        for (int kx = 0; kx < 3; ++kx) {
            const int ix = x + kx - 1;
            if (ix < 0 || ix > 255) continue;
            const float* xp = X + ((size_t)(b * 256 + iy) * 256 + ix) * 96;
            const float* wp = Wt + ((size_t)(o * 96)) * 9 + ky * 3 + kx;
            for (int c = 0; c < 96; ++c) acc += xp[c] * wp[c * 9];
        }
    }
    Y[i] = acc;
}

// ===========================================================================
// Host orchestration
// ===========================================================================
namespace {
struct Arena {
    char*  base;
    size_t off;
    size_t cap;
    float* alloc(size_t nfloats) {
        const size_t bytes = ((nfloats * sizeof(float)) + 255) & ~(size_t)255;
        float* p = (float*)(base + off);
        off += bytes;
        return p;
    }
};
struct Blk {
    const float *n1w, *n1b, *qkv_w, *qkv_b, *rpb, *proj_w, *proj_b;
    const float *n2w, *n2b, *fc1_w, *fc1_b, *fc2_w, *fc2_b;
};
struct Ups {
    const float *p_c1, *p_a, *p_c2, *b_c1, *b_c1b, *b_a, *b_c2, *conv;
};
} // namespace

extern "C" void kernel_launch(void* const* d_in, const int* in_sizes, int n_in,
                              void* d_out, int out_size, void* d_ws, size_t ws_size,
                              hipStream_t stream)
{
    (void)in_sizes; (void)n_in; (void)out_size;
    const int Bn = 4;
    const int HEADS[4] = {3, 6, 12, 24};

    // ---- walk flattened inputs in setup_inputs() insertion order ----------
    int cur = 0;
    auto F = [&]() { return (const float*)d_in[cur++]; };

    const float* x_in         = F();
    const float* conv_first_w = F();
    const float* conv_first_b = F();
    const float* pe_w  = F();
    const float* pe_b  = F();
    const float* pe_nw = F();
    const float* pe_nb = F();

    auto rdBlk = [&]() {
        Blk k;
        k.n1w = F(); k.n1b = F(); k.qkv_w = F(); k.qkv_b = F(); k.rpb = F();
        k.proj_w = F(); k.proj_b = F(); k.n2w = F(); k.n2b = F();
        k.fc1_w = F(); k.fc1_b = F(); k.fc2_w = F(); k.fc2_b = F();
        return k;
    };
    auto rdUps = [&]() {
        Ups u;
        u.p_c1 = F(); u.p_a = F(); u.p_c2 = F();
        u.b_c1 = F(); u.b_c1b = F(); u.b_a = F(); u.b_c2 = F(); u.conv = F();
        return u;
    };

    Blk encBlk[4][2];
    const float *mrg_nw[3], *mrg_nb[3], *mrg_red[3];
    for (int i = 0; i < 4; ++i) {
        for (int j = 0; j < 2; ++j) encBlk[i][j] = rdBlk();
        if (i < 3) { mrg_nw[i] = F(); mrg_nb[i] = F(); mrg_red[i] = F(); }
    }
    const float* norm_w = F();
    const float* norm_b = F();
    Ups up0 = rdUps();
    const float* dec_cat_w[3]; const float* dec_cat_b[3];
    Blk decBlk[3][2];
    Ups decUp[2];
    for (int k = 0; k < 3; ++k) {
        dec_cat_w[k] = F(); dec_cat_b[k] = F();
        for (int j = 0; j < 2; ++j) decBlk[k][j] = rdBlk();
        if (k < 2) decUp[k] = rdUps();
    }
    const float* norm_up_w = F();
    const float* norm_up_b = F();
    Ups up4 = rdUps();
    const float* out_w = F();

    // ---- helpers ----------------------------------------------------------
    Arena ar{(char*)d_ws, 0, ws_size};

    auto gemm = [&](const float* A, const float* Wt, const float* bias,
                    const float* res, const float* actp, float* C,
                    int M, int N, int K, int act) {
        dim3 g((N + 31) / 32, (M + 127) / 128), bl(32, 4);
        if (act == ACT_GELU)
            gemm_wmma<ACT_GELU><<<g, bl, 0, stream>>>(A, Wt, bias, res, actp, C, M, N, K);
        else if (act == ACT_PRELU)
            gemm_wmma<ACT_PRELU><<<g, bl, 0, stream>>>(A, Wt, bias, res, actp, C, M, N, K);
        else
            gemm_wmma<ACT_NONE><<<g, bl, 0, stream>>>(A, Wt, bias, res, actp, C, M, N, K);
    };
    auto ln = [&](const float* X, const float* g, const float* be, float* Y,
                  int rows, int C) {
        layernorm_k<<<(rows + 7) / 8, 256, 0, stream>>>(X, g, be, Y, rows, C);
    };
    auto elems = [](size_t n) { return (unsigned)((n + 255) / 256); };

    // one swin transformer block: x -> xout (both persistent slots)
    auto swin = [&](const Blk& p, const float* x, float* xout,
                    int H, int W, int C, int heads, int shift) {
        const int M = Bn * H * W;
        const int nW = (H / 8) * (W / 8);
        const int WH = Bn * nW * heads;
        const size_t mk = ar.off;
        float* lnb  = ar.alloc((size_t)M * C);
        float* qkvb = ar.alloc((size_t)M * 3 * C);
        float* sc   = ar.alloc((size_t)WH * 64 * 64);
        float* ao   = ar.alloc((size_t)M * C);
        float* x2   = ar.alloc((size_t)M * C);
        float* hb   = ar.alloc((size_t)M * 4 * C);

        ln(x, p.n1w, p.n1b, lnb, M, C);
        gemm(lnb, p.qkv_w, p.qkv_b, nullptr, nullptr, qkvb, M, 3 * C, C, ACT_NONE);
        attn_scores<<<WH, 256, 0, stream>>>(qkvb, p.rpb, sc, H, W, C, heads, shift, nW);
        softmax64<<<(WH * 64 + 7) / 8, 256, 0, stream>>>(sc, WH * 64);
        attn_av<<<WH, 256, 0, stream>>>(qkvb, sc, ao, H, W, C, heads, shift, nW);
        gemm(ao, p.proj_w, p.proj_b, /*res=*/x, nullptr, x2, M, C, C, ACT_NONE);
        ln(x2, p.n2w, p.n2b, lnb, M, C);
        gemm(lnb, p.fc1_w, p.fc1_b, nullptr, nullptr, hb, M, 4 * C, C, ACT_GELU);
        gemm(hb, p.fc2_w, p.fc2_b, /*res=*/x2, nullptr, xout, M, C, 4 * C, ACT_NONE);
        ar.off = mk;
    };

    // dual upsample (pixel-shuffle + bilinear branches)
    auto upsample = [&](const Ups& u, const float* xin, int H, int W, int C,
                        int factor, float* out) {
        const int M  = Bn * H * W;
        const int Co = (factor == 2) ? C / 2 : C;
        const int Ct = (factor == 4) ? 16 * C : 2 * C;
        const int M2 = M * factor * factor;
        const size_t mk = ar.off;
        float* xp = ar.alloc((size_t)M2 * Co);
        float* xb = ar.alloc((size_t)M2 * Co);
        {   // pixel-shuffle branch
            const size_t mk2 = ar.off;
            float* t1 = ar.alloc((size_t)M * Ct);
            float* t2 = ar.alloc((size_t)M2 * Co);
            gemm(xin, u.p_c1, nullptr, nullptr, u.p_a, t1, M, Ct, C, ACT_PRELU);
            pixel_shuffle_k<<<elems((size_t)M2 * Co), 256, 0, stream>>>(t1, t2, Bn, H, W, Co, factor);
            gemm(t2, u.p_c2, nullptr, nullptr, nullptr, xp, M2, Co, Co, ACT_NONE);
            ar.off = mk2;
        }
        {   // bilinear branch
            const size_t mk2 = ar.off;
            float* bb   = ar.alloc((size_t)M * C);
            float* bbig = ar.alloc((size_t)M2 * C);
            gemm(xin, u.b_c1, u.b_c1b, nullptr, u.b_a, bb, M, C, C, ACT_PRELU);
            bilinear_k<<<elems((size_t)M2 * C), 256, 0, stream>>>(bb, bbig, Bn, H, W, C, factor);
            gemm(bbig, u.b_c2, nullptr, nullptr, nullptr, xb, M2, Co, C, ACT_NONE);
            ar.off = mk2;
        }
        float* cat = ar.alloc((size_t)M2 * 2 * Co);
        concat_k<<<elems((size_t)M2 * 2 * Co), 256, 0, stream>>>(xp, xb, cat, (size_t)M2, Co, Co);
        gemm(cat, u.conv, nullptr, nullptr, nullptr, out, M2, Co, 2 * Co, ACT_NONE);
        ar.off = mk;
    };

    // ---- persistent buffers ------------------------------------------------
    const size_t SLOT = (size_t)Bn * 4096 * 96;     // M*C is constant per stage
    float* xA = ar.alloc(SLOT);
    float* xB = ar.alloc(SLOT);
    float* xds[3];
    xds[0] = ar.alloc((size_t)Bn * 4096 * 96);
    xds[1] = ar.alloc((size_t)Bn * 1024 * 192);
    xds[2] = ar.alloc((size_t)Bn * 256 * 384);

    float* x    = xA;
    float* xalt = xB;

    // ---- stem: conv_first -> patch embed -> LN ----------------------------
    {
        const size_t mk = ar.off;
        float* conv1 = ar.alloc((size_t)Bn * 96 * 256 * 256);
        float* tok   = ar.alloc((size_t)Bn * 4096 * 96);
        conv3x3_first<<<elems((size_t)Bn * 96 * 65536), 256, 0, stream>>>(
            x_in, conv_first_w, conv_first_b, conv1);
        patch_embed<<<elems((size_t)Bn * 4096 * 96), 256, 0, stream>>>(
            conv1, pe_w, pe_b, tok);
        ln(tok, pe_nw, pe_nb, x, Bn * 4096, 96);
        ar.off = mk;
    }

    // ---- encoder -----------------------------------------------------------
    int H = 64, W = 64, C = 96;
    for (int i = 0; i < 4; ++i) {
        const int M = Bn * H * W;
        if (i < 3)
            hipMemcpyAsync(xds[i], x, (size_t)M * C * sizeof(float),
                           hipMemcpyDeviceToDevice, stream);
        for (int j = 0; j < 2; ++j) {
            const int shift = (j == 1 && H > 8) ? 4 : 0;
            swin(encBlk[i][j], x, xalt, H, W, C, HEADS[i], shift);
            float* t = x; x = xalt; xalt = t;
        }
        if (i < 3) {
            const size_t mk = ar.off;
            const int rows = Bn * (H / 2) * (W / 2);
            float* mg   = ar.alloc((size_t)rows * 4 * C);
            float* mgln = ar.alloc((size_t)rows * 4 * C);
            merge_gather<<<elems((size_t)rows * 4 * C), 256, 0, stream>>>(x, mg, Bn, H, W, C);
            ln(mg, mrg_nw[i], mrg_nb[i], mgln, rows, 4 * C);
            gemm(mgln, mrg_red[i], nullptr, nullptr, nullptr, xalt, rows, 2 * C, 4 * C, ACT_NONE);
            { float* t = x; x = xalt; xalt = t; }
            H /= 2; W /= 2; C *= 2;
            ar.off = mk;
        }
    }

    // ---- bottleneck norm + up0 --------------------------------------------
    ln(x, norm_w, norm_b, xalt, Bn * H * W, C);
    { float* t = x; x = xalt; xalt = t; }
    upsample(up0, x, H, W, C, 2, xalt);          // (B, 16*16, 384)
    { float* t = x; x = xalt; xalt = t; }
    H *= 2; W *= 2; C /= 2;

    // ---- decoder -----------------------------------------------------------
    for (int k = 0; k < 3; ++k) {
        const int M = Bn * H * W;
        {
            const size_t mk = ar.off;
            float* cat = ar.alloc((size_t)M * 2 * C);
            concat_k<<<elems((size_t)M * 2 * C), 256, 0, stream>>>(
                x, xds[2 - k], cat, (size_t)M, C, C);
            gemm(cat, dec_cat_w[k], dec_cat_b[k], nullptr, nullptr, xalt, M, C, 2 * C, ACT_NONE);
            { float* t = x; x = xalt; xalt = t; }
            ar.off = mk;
        }
        for (int j = 0; j < 2; ++j) {
            const int shift = (j == 1 && H > 8) ? 4 : 0;
            swin(decBlk[k][j], x, xalt, H, W, C, HEADS[2 - k], shift);
            float* t = x; x = xalt; xalt = t;
        }
        if (k < 2) {
            upsample(decUp[k], x, H, W, C, 2, xalt);
            { float* t = x; x = xalt; xalt = t; }
            H *= 2; W *= 2; C /= 2;
        }
    }

    // ---- final norm, up4, output conv -------------------------------------
    ln(x, norm_up_w, norm_up_b, xalt, Bn * H * W, C);
    { float* t = x; x = xalt; xalt = t; }
    {
        const size_t mk = ar.off;
        float* xo = ar.alloc((size_t)Bn * 65536 * 96);   // (B,256,256,96) channel-last
        upsample(up4, x, 64, 64, 96, 4, xo);
        conv3x3_out<<<elems((size_t)Bn * 3 * 65536), 256, 0, stream>>>(
            xo, out_w, (float*)d_out);
        ar.off = mk;
    }
}
